// MultiheadAttention_56435870270139
// MI455X (gfx1250) — matrix operand order repaired and checked by reading, not compiled, not run on hardware
//
#include <hip/hip_runtime.h>

// Problem constants (from reference): N=2048, B=4, E=256, H=8, DH=32
#define EDIM 256
#define BDIM 4
#define SEQ  2048
#define HN   8
#define DHD  32
#define NTOK (SEQ * BDIM)      // 8192 tokens
#define BE   (BDIM * EDIM)     // 1024 elements per sequence-row stride

typedef __attribute__((ext_vector_type(16))) __bf16 v16bf;
typedef __attribute__((ext_vector_type(8)))  float  v8f;

#if defined(__gfx1250__) && __has_builtin(__builtin_amdgcn_tensor_load_to_lds)
#define CDNA5_HAS_TDM 1
#else
#define CDNA5_HAS_TDM 0
#endif

static __device__ __forceinline__ __bf16 f2bf(float f) {
  unsigned u = __float_as_uint(f);
  unsigned r = (u + 0x7FFFu + ((u >> 16) & 1u)) >> 16;  // RNE
  unsigned short h = (unsigned short)r;
  __bf16 b;
  __builtin_memcpy(&b, &h, 2);
  return b;
}

#if CDNA5_HAS_TDM
// Issue a 2D TDM tile load: tile[tile1 rows][tile0 elems] of bf16 from
// global (row stride = stride0 elems) into LDS at lds_off. The TDM pad
// engine inserts 4 DWORDs after every 16 DWORDs -> LDS row stride of
// (tile0=32) 32+8=40 bf16, matching the padded __shared__ tiles below.
static __device__ __forceinline__ void tdm_load_2d(
    unsigned lds_off, const __bf16* gptr,
    unsigned tdim0, unsigned tdim1,         // tensor extents (OOB clamp)
    unsigned tile0, unsigned tile1,         // tile extents (elements, rows)
    unsigned stride0)                       // tensor dim0 stride (elements)
{
  typedef __attribute__((ext_vector_type(4))) unsigned v4u;
  typedef __attribute__((ext_vector_type(4))) int      v4i;
  typedef __attribute__((ext_vector_type(8))) int      v8i;
  unsigned long long ga = (unsigned long long)gptr;   // byte address
  v4u g0;
  g0[0] = 1u;                                   // count=1, user descriptor
  g0[1] = lds_off;                              // lds_addr (bytes)
  g0[2] = (unsigned)ga;                         // global_addr[31:0]
  g0[3] = (unsigned)(ga >> 32) | (2u << 30);    // global_addr[56:32] | type=2
  v8i g1;
  // data_size=2B, pad_enable, pad_interval=3 (16 DW), pad_amount=3 (4 DW)
  g1[0] = (int)((1u << 16) | (1u << 20) | (3u << 22) | (3u << 25));
  g1[1] = (int)(tdim0 << 16);                        // tensor_dim0 lo16
  g1[2] = (int)((tdim0 >> 16) | (tdim1 << 16));      // dim0 hi | dim1 lo
  g1[3] = (int)((tdim1 >> 16) | (tile0 << 16));      // dim1 hi | tile_dim0
  g1[4] = (int)(tile1 & 0xFFFFu);                    // tile_dim1 (tile2=0)
  g1[5] = (int)stride0;                              // dim0_stride lo32
  g1[6] = 0;
  g1[7] = 0;
  v4i z = {0, 0, 0, 0};                              // 2D: groups 2/3 unused
#if __clang_major__ >= 23
  v8i z8 = {0, 0, 0, 0, 0, 0, 0, 0};
  __builtin_amdgcn_tensor_load_to_lds(g0, g1, z, z, z8, 0);
#else
  __builtin_amdgcn_tensor_load_to_lds(g0, g1, z, z, 0);
#endif
}
static __device__ __forceinline__ unsigned lds_addr_of(const void* p) {
  return (unsigned)(unsigned long long)p;   // generic LDS ptr: low 32b = offset
}
#endif

// ---------------------------------------------------------------- elementwise
__global__ __launch_bounds__(256) void k_addcvt(const float* __restrict__ a,
                                                const float* __restrict__ b,
                                                __bf16* __restrict__ o, int n) {
  int i = blockIdx.x * 256 + threadIdx.x;
  if (i < n) o[i] = f2bf(a[i] + b[i]);
}

__global__ __launch_bounds__(256) void k_cvt(const float* __restrict__ a,
                                             __bf16* __restrict__ o, int n) {
  int i = blockIdx.x * 256 + threadIdx.x;
  if (i < n) o[i] = f2bf(a[i]);
}

// ---------------------------------------------------------------- GEMM 256-K
// C[M,256] = A[M,256] @ W[256,256] + bias (+add0), M = 8192.
// Block: 256 threads = 8 waves; block tile 128(M) x 64(N); K staged 32 at a
// time. A panel arrives via TDM DMA (padded rows); W panel is staged
// transposed by VALU so each B fragment is one 32B LDS read.
__global__ __launch_bounds__(256) void k_gemm256(
    const __bf16* __restrict__ A,    // [M,256] bf16
    const __bf16* __restrict__ W,    // [256,256] bf16 row-major (k-major)
    const float*  __restrict__ bias, // [256]
    const float*  __restrict__ add0, // [M,256] or null
    float*        __restrict__ outF, // [M,256] or null
    __bf16*       __restrict__ outB) // [M,256] or null
{
  __shared__ __align__(32) __bf16 As[128][40];   // [m][k], 40 = TDM-padded row
  __shared__ __align__(32) __bf16 BsT[64][48];   // [n][k] transposed

  const int t    = threadIdx.x;
  const int wv   = t >> 5;
  const int lane = t & 31;
  const int lm   = lane & 15;
  const int hh   = lane >> 4;
  const int m0   = blockIdx.y * 128;
  const int n0   = blockIdx.x * 64;

  v8f acc[4] = {};

  for (int k0 = 0; k0 < EDIM; k0 += 32) {
    __syncthreads();
#if CDNA5_HAS_TDM
    // ---- A tile 128x32 via Tensor Data Mover (one wave issues the DMA)
    if (wv == 0) {
      tdm_load_2d(lds_addr_of(&As[0][0]),
                  A + (size_t)m0 * EDIM + k0,
                  EDIM, NTOK, 32, 128, EDIM);
    }
#else
    // ---- A tile 128x32 bf16 = 512 x uint4, 2 per thread
#pragma unroll
    for (int q = 0; q < 2; ++q) {
      int idx = t + q * 256;
      int row = idx >> 2;
      int c   = idx & 3;
      const __bf16* src = A + (size_t)(m0 + row) * EDIM + k0 + c * 8;
      *(uint4*)&As[row][c * 8] = *(const uint4*)src;
    }
#endif
    // ---- B tile 32x64 loaded transposed into BsT[n][k] (overlaps the DMA)
    {
      int row = t >> 3;          // k within chunk, 0..31
      int c   = t & 7;           // 8-col group
      const __bf16* src = W + (size_t)(k0 + row) * EDIM + n0 + c * 8;
      uint4 v = *(const uint4*)src;
      if (k0 + 32 < EDIM)
        __builtin_prefetch(src + 32 * EDIM, 0, 3);   // next W panel
      __bf16 tmp[8];
      __builtin_memcpy(tmp, &v, 16);
#pragma unroll
      for (int e = 0; e < 8; ++e) BsT[c * 8 + e][row] = tmp[e];
    }
#if CDNA5_HAS_TDM
    if (wv == 0) __builtin_amdgcn_s_wait_tensorcnt(0);
#endif
    __syncthreads();

    // ---- A fragment: 16x32, contiguous K per half (K 0-15 vs 16-31)
    v16bf af;
    {
      const __bf16* ar = &As[(wv << 4) + lm][hh << 4];
      uint4 lo = *(const uint4*)ar;
      uint4 hi = *(const uint4*)(ar + 8);
      __builtin_memcpy(&af, &lo, 16);
      __builtin_memcpy((char*)&af + 16, &hi, 16);
    }
    // ---- 4 B fragments + WMMAs
#pragma unroll
    for (int nt = 0; nt < 4; ++nt) {
      v16bf bf = *(const v16bf*)&BsT[(nt << 4) + lm][hh << 4];
      acc[nt] = __builtin_amdgcn_wmma_f32_16x16x32_bf16(
          false, af, false, bf, (short)0, acc[nt], false, false);
    }
  }

  // ---- epilogue (C layout: VGPR r -> row r + 8*half, col = lane&15)
#pragma unroll
  for (int nt = 0; nt < 4; ++nt) {
    int col = n0 + (nt << 4) + lm;
    float bv = bias[col];
#pragma unroll
    for (int r = 0; r < 8; ++r) {
      int row = m0 + (wv << 4) + (hh << 3) + r;
      size_t o = (size_t)row * EDIM + col;
      float v = acc[nt][r] + bv;
      if (add0) v += add0[o];
      if (outF) outF[o] = v;
      if (outB) outB[o] = f2bf(v);
    }
  }
}

// ---------------------------------------------------------------- attention
// One wave per (16 query rows, b, h). Flash-style online softmax over SEQ keys
// in chunks of 32. K/V panels are double-buffered in LDS via TDM; the
// in-order TENSORcnt (wait <=2) overlaps next-chunk DMA with current WMMAs.
__global__ __launch_bounds__(32) void k_attn(
    const __bf16* __restrict__ Q,    // [N,B,E] bf16
    const __bf16* __restrict__ K,
    const __bf16* __restrict__ V,    // vs
    const float*  __restrict__ VR,   // vr (f32, already in d_out)
    __bf16* __restrict__ CTX)        // [N,B,E] bf16 out
{
  __shared__ __align__(32) __bf16 Pl[16][40];      // P staging (C->A transpose)
#if CDNA5_HAS_TDM
  __shared__ __align__(32) __bf16 Kl[2][32][40];   // [buf][key][dh], padded
  __shared__ __align__(32) __bf16 Vl[2][32][40];
#endif

  const int lane = threadIdx.x;
  const int lm = lane & 15;
  const int hh = lane >> 4;
  const int r0 = blockIdx.x << 4;       // query row tile
  const int b  = blockIdx.y >> 3;       // H = 8
  const int h  = blockIdx.y & 7;
  const int c0 = b * EDIM + h * DHD;    // column base within [B*E] row

  // Persistent Q fragment (16 rows x DH=32 = exactly one WMMA K step)
  v16bf qf;
  {
    const __bf16* qp = Q + (size_t)(r0 + lm) * BE + c0 + (hh << 4);
    uint4 lo = *(const uint4*)qp;
    uint4 hi = *(const uint4*)(qp + 8);
    __builtin_memcpy(&qf, &lo, 16);
    __builtin_memcpy((char*)&qf + 16, &hi, 16);
  }

  const float scale = 0.1767766952966369f;   // 1/sqrt(32)
  float mrun[8], lrun[8];
  v8f acc0 = {}, acc1 = {};
#pragma unroll
  for (int r = 0; r < 8; ++r) { mrun[r] = -1e30f; lrun[r] = 0.f; }

#if CDNA5_HAS_TDM
  // preload chunk 0 (K then V; TENSORcnt is in-order per wave)
  tdm_load_2d(lds_addr_of(&Kl[0][0][0]), K + c0, BE, SEQ, 32, 32, BE);
  tdm_load_2d(lds_addr_of(&Vl[0][0][0]), V + c0, BE, SEQ, 32, 32, BE);
#endif

  int it = 0;
  for (int j0 = 0; j0 < SEQ; j0 += 32, ++it) {
    const int buf = it & 1;
#if CDNA5_HAS_TDM
    if (j0 + 32 < SEQ) {
      // kick next chunk's DMA, then wait only for the current chunk (<=2 left)
      tdm_load_2d(lds_addr_of(&Kl[buf ^ 1][0][0]),
                  K + (size_t)(j0 + 32) * BE + c0, BE, SEQ, 32, 32, BE);
      tdm_load_2d(lds_addr_of(&Vl[buf ^ 1][0][0]),
                  V + (size_t)(j0 + 32) * BE + c0, BE, SEQ, 32, 32, BE);
      __builtin_amdgcn_s_wait_tensorcnt(2);
    } else {
      __builtin_amdgcn_s_wait_tensorcnt(0);
    }
#endif
    // ---- scores for 32 keys: two WMMAs (K fragment = K^T, contiguous/lane)
    v8f s0, s1;
    {
#if CDNA5_HAS_TDM
      const __bf16* kp = &Kl[buf][lm][hh << 4];
#else
      const __bf16* kp = K + (size_t)(j0 + lm) * BE + c0 + (hh << 4);
#endif
      v16bf kf;
      uint4 lo = *(const uint4*)kp;
      uint4 hi = *(const uint4*)(kp + 8);
      __builtin_memcpy(&kf, &lo, 16);
      __builtin_memcpy((char*)&kf + 16, &hi, 16);
      v8f z = {};
      s0 = __builtin_amdgcn_wmma_f32_16x16x32_bf16(
          false, qf, false, kf, (short)0, z, false, false);
    }
    {
#if CDNA5_HAS_TDM
      const __bf16* kp = &Kl[buf][16 + lm][hh << 4];
#else
      const __bf16* kp = K + (size_t)(j0 + 16 + lm) * BE + c0 + (hh << 4);
#endif
      v16bf kf;
      uint4 lo = *(const uint4*)kp;
      uint4 hi = *(const uint4*)(kp + 8);
      __builtin_memcpy(&kf, &lo, 16);
      __builtin_memcpy((char*)&kf + 16, &hi, 16);
      v8f z = {};
      s1 = __builtin_amdgcn_wmma_f32_16x16x32_bf16(
          false, qf, false, kf, (short)0, z, false, false);
    }
    // ---- online softmax (row-wise reductions across each 16-lane half)
    float p0[8], p1[8];
#pragma unroll
    for (int r = 0; r < 8; ++r) {
      float a = s0[r] * scale, c = s1[r] * scale;
      float mx = fmaxf(a, c);
#pragma unroll
      for (int off = 1; off < 16; off <<= 1)
        mx = fmaxf(mx, __shfl_xor(mx, off, 32));
      float mn = fmaxf(mrun[r], mx);
      float al = __expf(mrun[r] - mn);
      float e0 = __expf(a - mn), e1 = __expf(c - mn);
      float rs = e0 + e1;
#pragma unroll
      for (int off = 1; off < 16; off <<= 1)
        rs += __shfl_xor(rs, off, 32);
      lrun[r] = lrun[r] * al + rs;
      mrun[r] = mn;
      acc0[r] *= al;
      acc1[r] *= al;
      p0[r] = e0;
      p1[r] = e1;
    }
    // ---- P: C layout -> LDS -> A layout fragment
#pragma unroll
    for (int r = 0; r < 8; ++r) {
      int row = (hh << 3) + r;
      Pl[row][lm]      = f2bf(p0[r]);
      Pl[row][16 + lm] = f2bf(p1[r]);
    }
    __syncthreads();
    v16bf pf;
    {
      const __bf16* pp = &Pl[lm][hh << 4];
      uint4 lo = *(const uint4*)pp;
      uint4 hi = *(const uint4*)(pp + 8);
      __builtin_memcpy(&pf, &lo, 16);
      __builtin_memcpy((char*)&pf + 16, &hi, 16);
    }
    // ---- ctx += P[16x32] @ V[32x32] : two WMMAs (dh halves)
#pragma unroll
    for (int dt = 0; dt < 2; ++dt) {
      v16bf vf;
#pragma unroll
      for (int e = 0; e < 16; ++e) {
#if CDNA5_HAS_TDM
        vf[e] = Vl[buf][(hh << 4) + e][(dt << 4) + lm];
#else
        int key = j0 + (hh << 4) + e;
        vf[e] = V[(size_t)key * BE + c0 + (dt << 4) + lm];
#endif
      }
      if (dt == 0)
        acc0 = __builtin_amdgcn_wmma_f32_16x16x32_bf16(
            false, pf, false, vf, (short)0, acc0, false, false);
      else
        acc1 = __builtin_amdgcn_wmma_f32_16x16x32_bf16(
            false, pf, false, vf, (short)0, acc1, false, false);
    }
    __syncthreads();
  }

  // ---- epilogue: normalize, add vr, store ctx (bf16) for output projection
#pragma unroll
  for (int r = 0; r < 8; ++r) {
    int row = r0 + (hh << 3) + r;
    float inv = 1.f / lrun[r];
    size_t base = (size_t)row * BE + c0;
    float v0 = acc0[r] * inv + VR[base + lm];
    float v1 = acc1[r] * inv + VR[base + 16 + lm];
    CTX[base + lm]      = f2bf(v0);
    CTX[base + 16 + lm] = f2bf(v1);
  }
}

// ---------------------------------------------------------------- launch
extern "C" void kernel_launch(void* const* d_in, const int* in_sizes, int n_in,
                              void* d_out, int out_size, void* d_ws, size_t ws_size,
                              hipStream_t stream) {
  (void)in_sizes; (void)n_in; (void)out_size; (void)ws_size;

  const float* x    = (const float*)d_in[0];
  const float* qpos = (const float*)d_in[1];
  const float* rres = (const float*)d_in[2];
  const float* sres = (const float*)d_in[3];
  const float* Wq = (const float*)d_in[4];  const float* bq = (const float*)d_in[5];
  const float* Wk = (const float*)d_in[6];  const float* bk = (const float*)d_in[7];
  const float* Wr = (const float*)d_in[8];  const float* br = (const float*)d_in[9];
  const float* Ws = (const float*)d_in[10]; const float* bs = (const float*)d_in[11];
  const float* Wo = (const float*)d_in[12]; const float* bo = (const float*)d_in[13];

  float* out0   = (float*)d_out;                       // x + attn_out
  float* out_vr = out0 + (size_t)NTOK * EDIM;          // vr
  float* out_vs = out0 + 2 * (size_t)NTOK * EDIM;      // vs

  char* p = (char*)d_ws;
  const size_t tokB = (size_t)NTOK * EDIM * sizeof(__bf16);  // 4 MiB each
  __bf16* qin  = (__bf16*)p; p += tokB;
  __bf16* qbf  = (__bf16*)p; p += tokB;
  __bf16* kbf  = (__bf16*)p; p += tokB;
  __bf16* vsbf = (__bf16*)p; p += tokB;
  __bf16* ctxb = (__bf16*)p; p += tokB;
  const size_t wB = (size_t)EDIM * EDIM * sizeof(__bf16);    // 128 KiB each
  __bf16* Wqb = (__bf16*)p; p += wB;
  __bf16* Wkb = (__bf16*)p; p += wB;
  __bf16* Wrb = (__bf16*)p; p += wB;
  __bf16* Wsb = (__bf16*)p; p += wB;
  __bf16* Wob = (__bf16*)p; p += wB;

  const int ntok = NTOK * EDIM;   // 2,097,152
  const int nw   = EDIM * EDIM;   // 65,536

  // q_in = x + query_pos  (bf16)
  k_addcvt<<<dim3(ntok / 256), dim3(256), 0, stream>>>(x, qpos, qin, ntok);
  // weights -> bf16
  k_cvt<<<dim3(nw / 256), dim3(256), 0, stream>>>(Wq, Wqb, nw);
  k_cvt<<<dim3(nw / 256), dim3(256), 0, stream>>>(Wk, Wkb, nw);
  k_cvt<<<dim3(nw / 256), dim3(256), 0, stream>>>(Wr, Wrb, nw);
  k_cvt<<<dim3(nw / 256), dim3(256), 0, stream>>>(Ws, Wsb, nw);
  k_cvt<<<dim3(nw / 256), dim3(256), 0, stream>>>(Wo, Wob, nw);

  dim3 gg(EDIM / 64, NTOK / 128), gb(256);
  // q, k projections (bf16 out for attention)
  k_gemm256<<<gg, gb, 0, stream>>>(qin, Wqb, bq, nullptr, nullptr, qbf);
  k_gemm256<<<gg, gb, 0, stream>>>(qin, Wkb, bk, nullptr, nullptr, kbf);
  // vr = q_in@Wr + br + receiver_val_res  -> output 1 (f32)
  k_gemm256<<<gg, gb, 0, stream>>>(qin, Wrb, br, rres, out_vr, nullptr);
  // vs = q_in@Ws + bs + sender_val_res    -> output 2 (f32) + bf16 for attention
  k_gemm256<<<gg, gb, 0, stream>>>(qin, Wsb, bs, sres, out_vs, vsbf);

  // flash attention per (row-tile, b*h); ctx = softmax(QK^T/sqrt(dh))@vs + vr
  k_attn<<<dim3(SEQ / 16, BDIM * HN), dim3(32), 0, stream>>>(
      qbf, kbf, vsbf, out_vr, ctxb);

  // out0 = x + ctx@Wo + bo
  k_gemm256<<<gg, gb, 0, stream>>>(ctxb, Wob, bo, x, out0, nullptr);
}
